// NormalSelfAttention1d_23390391894152
// MI455X (gfx1250) — matrix operand order repaired and checked by reading, not compiled, not run on hardware
//
#include <hip/hip_runtime.h>
#include <hip/hip_bf16.h>
#include <math.h>

// MI455X / gfx1250: wave32, WMMA 16x16x32 bf16.
// Pipeline:
//   1) xpose_kernel : x fp32 [B,D,N] -> xt bf16 [B,N,D] (LDS-tiled transpose)
//   2) wcvt_kernel  : Wq/Wk/Wv fp32 -> bf16 (one-time)
//   3) 3x qkv_proj  : bf16 WMMA GEMM -> Q,K [B,N,D] bf16 ; V^T [B,D,N] bf16
//   4) attn_kernel  : cooperative flash attention (4-wave WG, d-split)
// Workspace: xt 16 MiB + Q/K/VT 48 MiB + Wbf 1.5 MiB ~= 66 MiB.

typedef __attribute__((ext_vector_type(16))) __bf16 v16bf;
typedef __attribute__((ext_vector_type(8)))  __bf16 v8bf;
typedef __attribute__((ext_vector_type(8)))  float  v8f;
typedef __attribute__((ext_vector_type(4)))  float  v4f;

#define BATCH 8
#define DIM   512
#define SEQ   2048

__device__ __forceinline__ __bf16 f2bf(float f) { return (__bf16)f; }

__device__ __forceinline__ v8f wmma_bf16(v16bf a, v16bf b, v8f c) {
  return __builtin_amdgcn_wmma_f32_16x16x32_bf16(false, a, false, b, (short)0, c, false, false);
}

// ---------------------------------------------------------------------------
// x [B,D,N] fp32 -> xt [B,N,D] bf16, 32x32 tiles through LDS (conflict-free).
// ---------------------------------------------------------------------------
__global__ __launch_bounds__(256)
void xpose_kernel(const float* __restrict__ x, __bf16* __restrict__ xt)
{
  __shared__ __bf16 tile[32][33];
  int bi = blockIdx.z;
  int c0 = blockIdx.y * 32;     // D coordinate
  int n0 = blockIdx.x * 32;     // N coordinate
  int tx = threadIdx.x;         // 0..31
  int ty = threadIdx.y;         // 0..7

  const float* xb = x + (size_t)bi * DIM * SEQ;
  #pragma unroll
  for (int i = 0; i < 4; ++i) {
    int c = ty + 8 * i;
    tile[c][tx] = f2bf(xb[(size_t)(c0 + c) * SEQ + n0 + tx]);   // coalesced over n
  }
  __syncthreads();
  __bf16* xo = xt + (size_t)bi * SEQ * DIM;
  #pragma unroll
  for (int i = 0; i < 4; ++i) {
    int n = ty + 8 * i;
    xo[(size_t)(n0 + n) * DIM + c0 + tx] = tile[tx][n];         // coalesced over d
  }
}

// ---------------------------------------------------------------------------
// One-time fp32 -> bf16 weight conversion.
// ---------------------------------------------------------------------------
__global__ __launch_bounds__(256)
void wcvt_kernel(const float* __restrict__ W, __bf16* __restrict__ Wb, int n)
{
  int i = blockIdx.x * 256 + threadIdx.x;
  if (i < n) Wb[i] = f2bf(W[i]);
}

// ---------------------------------------------------------------------------
// QKV projection: Y[b,n,do] = sum_c xt[b,n,c] * W[do,c] + bias[do]
// One wave computes a 16(n) x 64(d) tile, all-bf16 inputs, fp32 accumulate.
// Output row-major [B,N,D] (transposed=0, Q/K) or [B,D,N] (transposed=1, V^T).
// ---------------------------------------------------------------------------
__global__ __launch_bounds__(32)
void qkv_proj_kernel(const __bf16* __restrict__ xt,   // [B, N, D] bf16
                     const __bf16* __restrict__ W,    // [D, D] bf16 row-major
                     const float*  __restrict__ bias, // [D] fp32
                     __bf16* __restrict__ out,
                     int transposed)
{
  const int NT = SEQ / 16, DQ = DIM / 64;
  int wid = blockIdx.x;                 // B * NT * DQ = 8192
  int b   = wid / (NT * DQ);
  int r   = wid % (NT * DQ);
  int n0  = (r / DQ) * 16;
  int dq  = (r % DQ) * 64;

  int lane  = threadIdx.x & 31;
  int col   = lane & 15;
  int hi    = lane >> 4;
  int kb0   = hi * 16;     // 16-bit A 16x32 per-lane K base (contiguous K)
  int khalf = hi * 16;     // 16-bit B 32x16 per-lane K base

  // A fragments: contiguous v8bf pairs from xt rows.
  const __bf16* Abase = xt + (size_t)b * SEQ * DIM + (size_t)(n0 + col) * DIM;
  v16bf af[DIM / 32];
  #pragma unroll
  for (int cc = 0; cc < DIM / 32; ++cc) {
    v8bf lo = *(const v8bf*)(Abase + cc * 32 + kb0);
    v8bf hh = *(const v8bf*)(Abase + cc * 32 + kb0 + 8);
    v16bf a;
    #pragma unroll
    for (int j = 0; j < 8; ++j) { a[j] = lo[j]; a[j + 8] = hh[j]; }
    af[cc] = a;
  }

  #pragma unroll
  for (int dt = 0; dt < 4; ++dt) {
    int d0 = dq + dt * 16;
    const __bf16* wbase = W + (size_t)(d0 + col) * DIM + khalf;  // contiguous K
    v8f acc = {};
    #pragma unroll
    for (int cc = 0; cc < DIM / 32; ++cc) {
      v16bf bf = *(const v16bf*)(wbase + cc * 32);
      acc = wmma_bf16(af[cc], bf, acc);
    }
    float bb = bias[d0 + col];
    #pragma unroll
    for (int rr = 0; rr < 8; ++rr) {
      float v  = acc[rr] + bb;
      int nrow = n0 + rr + hi * 8;
      if (!transposed)
        out[(size_t)b * SEQ * DIM + (size_t)nrow * DIM + d0 + col] = f2bf(v);
      else
        out[(size_t)b * DIM * SEQ + (size_t)(d0 + col) * SEQ + nrow] = f2bf(v);
    }
  }
}

// ---------------------------------------------------------------------------
// Cooperative flash attention. 128-thread WG (4 waves) per (b, 16-row tile).
// Wave w owns d-slice [128w, 128w+128): computes partial scores over it and
// accumulates the 16x128 O slice in registers (8 C fragments).
// ---------------------------------------------------------------------------
__global__ __launch_bounds__(128)
void attn_kernel(const __bf16* __restrict__ Qb,  // [B, N, D]
                 const __bf16* __restrict__ Kb,  // [B, N, D]
                 const __bf16* __restrict__ Vt,  // [B, D, N]
                 float* __restrict__ out)        // [B, D, N]
{
  __shared__ float  Ssh[512][4];                                    // 8 KB: [pos][wave]
  __shared__ __bf16 Psh[4][16 * 32] __attribute__((aligned(32)));   // 4 KB: per-wave P copy
  __shared__ float  Obuf[4][16 * 128];                              // 32 KB: per-wave O stage
  __shared__ float  Lsh[16];

  int wgid = blockIdx.x;                // B * N/16 = 1024
  int b    = wgid / (SEQ / 16);
  int n0   = (wgid % (SEQ / 16)) * 16;

  int tid   = threadIdx.x;
  int w     = tid >> 5;                 // wave id 0..3
  int lane  = tid & 31;
  int col   = lane & 15;
  int hi    = lane >> 4;
  int kb0   = hi * 16;
  int khalf = hi * 16;
  int dbase = w * 128;                  // this wave's d-slice origin

  const __bf16* Qbase = Qb + (size_t)b * SEQ * DIM + (size_t)(n0 + col) * DIM + dbase;
  const __bf16* Kbb   = Kb + (size_t)b * SEQ * DIM + dbase;
  const __bf16* Vbb   = Vt + (size_t)b * DIM * SEQ;

  // Q fragments for this wave's d-slice: 4 x v16bf = 32 VGPRs.
  v16bf qf[4];
  #pragma unroll
  for (int cc = 0; cc < 4; ++cc) {
    v8bf lo = *(const v8bf*)(Qbase + cc * 32 + kb0);
    v8bf hh = *(const v8bf*)(Qbase + cc * 32 + kb0 + 8);
    v16bf q;
    #pragma unroll
    for (int j = 0; j < 8; ++j) { q[j] = lo[j]; q[j + 8] = hh[j]; }
    qf[cc] = q;
  }

  // O slice accumulators: 8 tiles of 16 cols -> 64 VGPRs, resident in registers.
  v8f oacc[8];
  #pragma unroll
  for (int t = 0; t < 8; ++t) oacc[t] = (v8f){};

  float M[8], L[8], alpha[8];
  #pragma unroll
  for (int r = 0; r < 8; ++r) { M[r] = -__builtin_inff(); L[r] = 0.0f; }

  for (int m0 = 0; m0 < SEQ; m0 += 32) {
    // ---- partial scores over this wave's 128-wide d slice ----
    v8f s0p = {}, s1p = {};
    {
      const __bf16* k0 = Kbb + (size_t)(m0 + col) * DIM + khalf;
      const __bf16* k1 = Kbb + (size_t)(m0 + 16 + col) * DIM + khalf;
      #pragma unroll
      for (int cc = 0; cc < 4; ++cc) {
        v16bf kf0 = *(const v16bf*)(k0 + cc * 32);
        s0p = wmma_bf16(qf[cc], kf0, s0p);
        v16bf kf1 = *(const v16bf*)(k1 + cc * 32);
        s1p = wmma_bf16(qf[cc], kf1, s1p);
      }
      if (m0 + 32 < SEQ) {              // gfx1250 global_prefetch_b8
        __builtin_prefetch(k0 + (size_t)32 * DIM, 0, 1);
        __builtin_prefetch(k1 + (size_t)32 * DIM, 0, 1);
      }
    }

    // ---- publish partials: Ssh[pos][wave], wave-contiguous for v4 reduce ----
    #pragma unroll
    for (int r = 0; r < 8; ++r) {
      int prow = r + hi * 8;
      Ssh[prow * 16 + col][w]       = s0p[r];
      Ssh[256 + prow * 16 + col][w] = s1p[r];
    }
    __syncthreads();   // barrier A

    // ---- reduce across waves + online softmax (redundant in every wave) ----
    __bf16* Pme = &Psh[w][0];
    #pragma unroll
    for (int r = 0; r < 8; ++r) {
      int prow = r + hi * 8;
      v4f a0 = *(const v4f*)&Ssh[prow * 16 + col][0];
      v4f a1 = *(const v4f*)&Ssh[256 + prow * 16 + col][0];
      float sv0 = (a0[0] + a0[1]) + (a0[2] + a0[3]);
      float sv1 = (a1[0] + a1[1]) + (a1[2] + a1[3]);

      float mx = fmaxf(sv0, sv1);
      mx = fmaxf(mx, __shfl_xor(mx, 1, 32));
      mx = fmaxf(mx, __shfl_xor(mx, 2, 32));
      mx = fmaxf(mx, __shfl_xor(mx, 4, 32));
      mx = fmaxf(mx, __shfl_xor(mx, 8, 32));
      float Mn = fmaxf(M[r], mx);
      float av = __expf(M[r] - Mn);
      float p0 = __expf(sv0 - Mn);
      float p1 = __expf(sv1 - Mn);
      float rs = p0 + p1;
      rs += __shfl_xor(rs, 1, 32);
      rs += __shfl_xor(rs, 2, 32);
      rs += __shfl_xor(rs, 4, 32);
      rs += __shfl_xor(rs, 8, 32);
      L[r]     = L[r] * av + rs;
      M[r]     = Mn;
      alpha[r] = av;
      Pme[prow * 32 + col]      = f2bf(p0);
      Pme[prow * 32 + 16 + col] = f2bf(p1);
    }
    __syncthreads();   // barrier B: Ssh consumed, own Psh visible

    // ---- P tile: C-layout -> A-layout through own LDS copy ----
    v16bf pf;
    {
      v8bf plo = *(const v8bf*)(&Pme[col * 32 + kb0]);
      v8bf phi = *(const v8bf*)(&Pme[col * 32 + kb0 + 8]);
      #pragma unroll
      for (int j = 0; j < 8; ++j) { pf[j] = plo[j]; pf[j + 8] = phi[j]; }
    }

    // ---- O_slice = diag(alpha) * O_slice + P @ V_slice (registers only) ----
    #pragma unroll
    for (int t = 0; t < 8; ++t) {
      const __bf16* vp = Vbb + (size_t)(dbase + t * 16 + col) * SEQ + m0 + khalf;
      v16bf vf = *(const v16bf*)vp;
      v8f c = oacc[t];
      #pragma unroll
      for (int r = 0; r < 8; ++r) c[r] *= alpha[r];
      oacc[t] = wmma_bf16(pf, vf, c);
      if (m0 + 32 < SEQ) __builtin_prefetch(vp + 32, 0, 1);
    }
  }

  // ---- finalize: stage O slice in LDS, write [B,D,N] with 128-bit stores ----
  float* Ome = &Obuf[w][0];
  #pragma unroll
  for (int t = 0; t < 8; ++t)
    #pragma unroll
    for (int r = 0; r < 8; ++r)
      Ome[(r + hi * 8) * 128 + t * 16 + col] = oacc[t][r];

  if (w == 0 && col == 0) {
    #pragma unroll
    for (int r = 0; r < 8; ++r) Lsh[hi * 8 + r] = L[r];
  }
  __syncthreads();

  float invL[16];
  #pragma unroll
  for (int n = 0; n < 16; ++n) invL[n] = 1.0f / Lsh[n];

  float* ob = out + (size_t)b * DIM * SEQ + (size_t)n0;
  #pragma unroll
  for (int i = 0; i < 4; ++i) {
    int dd = lane + 32 * i;             // column within this wave's slice
    int d  = dbase + dd;
    float vals[16];
    #pragma unroll
    for (int n = 0; n < 16; ++n) vals[n] = Ome[n * 128 + dd] * invL[n];
    #pragma unroll
    for (int g = 0; g < 4; ++g) {
      v4f v;
      v[0] = vals[4 * g + 0]; v[1] = vals[4 * g + 1];
      v[2] = vals[4 * g + 2]; v[3] = vals[4 * g + 3];
      *(v4f*)(ob + (size_t)d * SEQ + 4 * g) = v;
    }
  }
}

// ---------------------------------------------------------------------------
extern "C" void kernel_launch(void* const* d_in, const int* in_sizes, int n_in,
                              void* d_out, int out_size, void* d_ws, size_t ws_size,
                              hipStream_t stream) {
  (void)in_sizes; (void)n_in; (void)out_size; (void)ws_size;
  const float* x  = (const float*)d_in[0];
  const float* Wq = (const float*)d_in[1];
  const float* bq = (const float*)d_in[2];
  const float* Wk = (const float*)d_in[3];
  const float* bk = (const float*)d_in[4];
  const float* Wv = (const float*)d_in[5];
  const float* bv = (const float*)d_in[6];

  const size_t elems = (size_t)BATCH * SEQ * DIM;   // 8.4M
  const size_t wlen  = (size_t)DIM * DIM;           // 262144
  __bf16* XT  = (__bf16*)d_ws;         // [B,N,D] bf16
  __bf16* Q   = XT + elems;            // [B,N,D]
  __bf16* K   = Q + elems;             // [B,N,D]
  __bf16* VT  = K + elems;             // [B,D,N]
  __bf16* Wqb = VT + elems;            // [D,D] bf16
  __bf16* Wkb = Wqb + wlen;
  __bf16* Wvb = Wkb + wlen;

  // 1) transpose+convert x, convert weights
  xpose_kernel<<<dim3(SEQ / 32, DIM / 32, BATCH), dim3(32, 8), 0, stream>>>(x, XT);
  wcvt_kernel<<<(int)(wlen / 256), 256, 0, stream>>>(Wq, Wqb, (int)wlen);
  wcvt_kernel<<<(int)(wlen / 256), 256, 0, stream>>>(Wk, Wkb, (int)wlen);
  wcvt_kernel<<<(int)(wlen / 256), 256, 0, stream>>>(Wv, Wvb, (int)wlen);

  // 2) projections
  int projGrid = BATCH * (SEQ / 16) * (DIM / 64);   // 8192 waves
  qkv_proj_kernel<<<projGrid, dim3(32), 0, stream>>>(XT, Wqb, bq, Q,  0);
  qkv_proj_kernel<<<projGrid, dim3(32), 0, stream>>>(XT, Wkb, bk, K,  0);
  qkv_proj_kernel<<<projGrid, dim3(32), 0, stream>>>(XT, Wvb, bv, VT, 1);

  // 3) attention
  int attnGrid = BATCH * (SEQ / 16);                // 1024 WGs x 4 waves
  attn_kernel<<<attnGrid, dim3(128), 0, stream>>>(Q, K, VT, (float*)d_out);
}
